// LBLbiLm_43731357007885
// MI455X (gfx1250) — matrix operand order repaired and checked by reading, not compiled, not run on hardware
//
#include <hip/hip_runtime.h>
#include <hip/hip_bf16.h>

// ---------------------------------------------------------------------------
// Problem constants (from reference): B=32, S=1024, H=512, W=4, L=2
// ---------------------------------------------------------------------------
#define Hc   512
#define Sc   1024
#define Bc   32
#define Lc   2
#define Mtot (Bc * Sc)          // 32768 rows
#define HH   (Hc * Hc)          // 262144 elements per weight matrix

// LDS tile geometry for the GEMMs (block tile 128M x 64N, K-step 32)
#define LDS_STRIDE 40           // padded row stride (f16 elems) to spread banks
#define ASZ (128 * LDS_STRIDE)  // one A buffer (f16 elems)
#define BSZ (64  * LDS_STRIDE)  // one B buffer (f16 elems)

typedef _Float16 f16;
typedef __attribute__((ext_vector_type(4)))  _Float16 v4h_t;
typedef __attribute__((ext_vector_type(8)))  _Float16 v8h_t;
typedef __attribute__((ext_vector_type(16))) _Float16 v16h_t;
typedef __attribute__((ext_vector_type(8)))  float    v8f_t;

// ---------------------------------------------------------------------------
// WMMA fragment loaders (per CDNA5 ISA 7.12.2 layouts, wave32)
//   A 16x32 f16:  lane m = lane&15, half = lane>>4
//     elems 0..7  : K = 8*half + 0..7
//     elems 8..15 : K = 16 + 8*half + 0..7
//   B 32x16 f16:  lane n = lane&15, half = lane>>4
//     elems 0..7  : K = 8*half + 0..7
//     elems 8..15 : K = 16 + 8*half + 0..7
// ---------------------------------------------------------------------------
__device__ __forceinline__ v16h_t load_a_frag(const f16* p) {
  v8h_t lo = *reinterpret_cast<const v8h_t*>(p);
  v8h_t hi = *reinterpret_cast<const v8h_t*>(p + 16);
  return __builtin_shufflevector(lo, hi, 0,1,2,3,4,5,6,7,8,9,10,11,12,13,14,15);
}
__device__ __forceinline__ v16h_t load_b_frag(const f16* p) {
  v8h_t lo = *reinterpret_cast<const v8h_t*>(p);
  v8h_t hi = *reinterpret_cast<const v8h_t*>(p + 16);
  return __builtin_shufflevector(lo, hi, 0,1,2,3,4,5,6,7,8,9,10,11,12,13,14,15);
}

#define WMMA_F16(A, Bf, C) \
  __builtin_amdgcn_wmma_f32_16x16x32_f16(false, (A), false, (Bf), (short)0, (C), false, false)

// ---------------------------------------------------------------------------
// CDNA5 async global->LDS copy (ASYNCcnt-tracked), GVS addressing:
//   lds[vdst + inst_off] = mem[saddr + vaddr]
// The LDS-address VGPR takes the workgroup-relative LDS byte offset, which is
// the low 32 bits of the flat shared-aperture pointer (ISA 10.2).
// ---------------------------------------------------------------------------
__device__ __forceinline__ void async_copy16(uint32_t lds_off, const void* base,
                                             uint32_t byte_off) {
  asm volatile("global_load_async_to_lds_b128 %0, %1, %2"
               :: "v"(lds_off), "v"(byte_off), "s"(base)
               : "memory");
}
__device__ __forceinline__ void wait_async0() {
  asm volatile("s_wait_asynccnt 0x0" ::: "memory");
}
__device__ __forceinline__ uint32_t lds_off_of(const void* p) {
  return (uint32_t)(uintptr_t)p;   // LDS aperture: addr[31:0] = LDS byte offset
}

// ---------------------------------------------------------------------------
// Shared GEMM mainloop: C(128x64 block tile) = A[M,512] @ Wt[512,512]^T,
// double-buffered async staging through LDS, 2x2 WMMA per wave (32x32 tile).
// ---------------------------------------------------------------------------
__device__ __forceinline__ void gemm_mainloop(const f16* __restrict__ A,
                                              const f16* __restrict__ Wt,
                                              f16* smA, f16* smB,
                                              v8f_t acc[2][2])
{
  const int t    = threadIdx.x;
  const int lane = t & 31;
  const int wave = t >> 5;
  const int hf   = lane >> 4;
  const int l16  = lane & 15;
  const int mblock = blockIdx.y * 128;
  const int nblock = blockIdx.x * 64;

  // --- async-copy ownership: thread t copies A rows (t>>2) and (t>>2)+64,
  //     col-chunk (t&3)*8, and B row (t>>2), same col-chunk. 16B each.
  const int ar = t >> 2;            // 0..63
  const int ac = (t & 3) * 8;       // element column within 32-wide K slab
  const uint32_t gA0 = ((uint32_t)(mblock + ar) * Hc + (uint32_t)ac) * 2u;
  const uint32_t gA1 = gA0 + 64u * Hc * 2u;
  const uint32_t gB  = ((uint32_t)(nblock + ar) * Hc + (uint32_t)ac) * 2u;

  uint32_t lA0[2], lA1[2], lB[2];
#pragma unroll
  for (int c = 0; c < 2; ++c) {
    lA0[c] = lds_off_of(&smA[c * ASZ + ar * LDS_STRIDE + ac]);
    lA1[c] = lds_off_of(&smA[c * ASZ + (ar + 64) * LDS_STRIDE + ac]);
    lB [c] = lds_off_of(&smB[c * BSZ + ar * LDS_STRIDE + ac]);
  }

  // --- per-wave fragment read pointers into LDS
  const f16* sArow = smA + ((wave & 3) * 32 + l16) * LDS_STRIDE + 8 * hf;
  const f16* sBrow = smB + ((wave >> 2) * 32 + l16) * LDS_STRIDE + 8 * hf;

  // prologue: stage K-slab 0 into buffer 0
  async_copy16(lA0[0], A, gA0);
  async_copy16(lA1[0], A, gA1);
  async_copy16(lB[0],  Wt, gB);

  for (int step = 0; step < 16; ++step) {
    const int cur = step & 1;
    wait_async0();        // my copies for buf[cur] have landed in LDS
    __syncthreads();      // everyone's copies landed; prev reads of buf[cur^1] done

    if (step < 15) {      // stage next K-slab into the other buffer
      const uint32_t ko = (uint32_t)(step + 1) * 32u * 2u;
      async_copy16(lA0[cur ^ 1], A, gA0 + ko);
      async_copy16(lA1[cur ^ 1], A, gA1 + ko);
      async_copy16(lB[cur ^ 1],  Wt, gB + ko);
    }

    const f16* aA = sArow + cur * ASZ;
    const f16* aB = sBrow + cur * BSZ;
    const v16h_t A0 = load_a_frag(aA);
    const v16h_t A1 = load_a_frag(aA + 16 * LDS_STRIDE);
    const v16h_t B0 = load_b_frag(aB);
    const v16h_t B1 = load_b_frag(aB + 16 * LDS_STRIDE);
    acc[0][0] = WMMA_F16(A0, B0, acc[0][0]);
    acc[0][1] = WMMA_F16(A0, B1, acc[0][1]);
    acc[1][0] = WMMA_F16(A1, B0, acc[1][0]);
    acc[1][1] = WMMA_F16(A1, B1, acc[1][1]);
  }
}

// ---------------------------------------------------------------------------
// Convert the 8 weight matrices (lw1[2], lw2[2], rw1[2], rw2[2]) to f16.
// dst layout: mat index = side*4 + (w2 ? 2 : 0) + layer
// ---------------------------------------------------------------------------
__global__ __launch_bounds__(256)
void wconv_kernel(const float* __restrict__ lw1, const float* __restrict__ lw2,
                  const float* __restrict__ rw1, const float* __restrict__ rw2,
                  f16* __restrict__ dst)
{
  const size_t idx = ((size_t)blockIdx.x * 256 + threadIdx.x) * 4;
  const int    mat = (int)(idx >> 18);
  const size_t rem = idx & (size_t)(HH - 1);
  const float* src;
  if      (mat < 2) src = lw1 + (size_t)(mat)     * HH;
  else if (mat < 4) src = lw2 + (size_t)(mat - 2) * HH;
  else if (mat < 6) src = rw1 + (size_t)(mat - 4) * HH;
  else              src = rw2 + (size_t)(mat - 6) * HH;
  const float4 v = *reinterpret_cast<const float4*>(src + rem);
  v4h_t o;
  o[0] = (f16)v.x; o[1] = (f16)v.y; o[2] = (f16)v.z; o[3] = (f16)v.w;
  *reinterpret_cast<v4h_t*>(dst + idx) = o;
}

// ---------------------------------------------------------------------------
// Windowed weighted sums -> resident activation x in out1 region.
// out1 row layout: [b*S + t] * 1024, cols 0..511 = left, 512..1023 = right.
// ---------------------------------------------------------------------------
__device__ __forceinline__ float4 padded_row4(const float* inp_b, const float* lp,
                                              const float* rp, int i, int h)
{
  const float* p;
  if (i < 4)         p = lp    + (size_t)i * Hc + h;
  else if (i < 1028) p = inp_b + (size_t)(i - 4) * Hc + h;
  else               p = rp    + (size_t)(i - 1028) * Hc + h;
  return *reinterpret_cast<const float4*>(p);
}

__global__ __launch_bounds__(256)
void window_kernel(const float* __restrict__ inputs, const float* __restrict__ lp,
                   const float* __restrict__ rp, const float* __restrict__ lwt,
                   const float* __restrict__ rwt, float* __restrict__ out1)
{
  const int tid = blockIdx.x * 256 + threadIdx.x;
  const int h = (tid & 127) * 4;
  const int t = (tid >> 7) & (Sc - 1);
  const int b = tid >> 17;
  const float* inp_b = inputs + (size_t)b * Sc * Hc;
  float4 la = {0.f, 0.f, 0.f, 0.f}, ra = {0.f, 0.f, 0.f, 0.f};
#pragma unroll
  for (int w = 0; w < 4; ++w) {
    const float lw = lwt[w], rw = rwt[w];
    const float4 pl = padded_row4(inp_b, lp, rp, t + w, h);
    const float4 pr = padded_row4(inp_b, lp, rp, t + w + 5, h);
    la.x += lw * pl.x; la.y += lw * pl.y; la.z += lw * pl.z; la.w += lw * pl.w;
    ra.x += rw * pr.x; ra.y += rw * pr.y; ra.z += rw * pr.z; ra.w += rw * pr.w;
  }
  float* orow = out1 + ((size_t)b * Sc + t) * (2 * Hc);
  *reinterpret_cast<float4*>(orow + h)      = la;
  *reinterpret_cast<float4*>(orow + Hc + h) = ra;
}

// ---------------------------------------------------------------------------
// LayerNorm (annotated-transformer style, unbiased std): one wave per row.
// ---------------------------------------------------------------------------
__global__ __launch_bounds__(256)
void ln_kernel(const float* __restrict__ xbase, const float* __restrict__ g,
               const float* __restrict__ bt, f16* __restrict__ Y)
{
  const int lane = threadIdx.x & 31;
  const int row  = blockIdx.x * 8 + (threadIdx.x >> 5);
  const float* xr = xbase + (size_t)row * (2 * Hc);
  float4 v[4];
  float s = 0.f, s2 = 0.f;
#pragma unroll
  for (int q = 0; q < 4; ++q) {
    v[q] = *reinterpret_cast<const float4*>(xr + q * 128 + lane * 4);
    s  += v[q].x + v[q].y + v[q].z + v[q].w;
    s2 += v[q].x * v[q].x + v[q].y * v[q].y + v[q].z * v[q].z + v[q].w * v[q].w;
  }
#pragma unroll
  for (int off = 16; off > 0; off >>= 1) {
    s  += __shfl_xor(s,  off, 32);
    s2 += __shfl_xor(s2, off, 32);
  }
  const float mu  = s * (1.f / 512.f);
  float var = (s2 * (1.f / 512.f) - mu * mu) * (512.f / 511.f);
  var = var > 0.f ? var : 0.f;
  const float rs = 1.f / (__builtin_sqrtf(var) + 1e-6f);
#pragma unroll
  for (int q = 0; q < 4; ++q) {
    const int col = q * 128 + lane * 4;
    const float4 gg = *reinterpret_cast<const float4*>(g + col);
    const float4 bb = *reinterpret_cast<const float4*>(bt + col);
    v4h_t o;
    o[0] = (f16)(gg.x * (v[q].x - mu) * rs + bb.x);
    o[1] = (f16)(gg.y * (v[q].y - mu) * rs + bb.y);
    o[2] = (f16)(gg.z * (v[q].z - mu) * rs + bb.z);
    o[3] = (f16)(gg.w * (v[q].w - mu) * rs + bb.w);
    *reinterpret_cast<v4h_t*>(Y + (size_t)row * Hc + col) = o;
  }
}

// ---------------------------------------------------------------------------
// GEMM 1: Out = relu(A @ Wt^T + bias), f16 output.
// ---------------------------------------------------------------------------
__global__ __launch_bounds__(256)
void gemm_relu_kernel(const f16* __restrict__ A, const f16* __restrict__ Wt,
                      const float* __restrict__ bias, f16* __restrict__ Out)
{
  __shared__ __align__(16) f16 smA[2 * ASZ];
  __shared__ __align__(16) f16 smB[2 * BSZ];

  const int lane  = threadIdx.x & 31;
  const int wave  = threadIdx.x >> 5;
  const int mbase = blockIdx.y * 128 + (wave & 3) * 32;
  const int nbase = blockIdx.x * 64  + (wave >> 2) * 32;
  const int hf  = lane >> 4;
  const int l16 = lane & 15;

  v8f_t acc[2][2] = {};
  gemm_mainloop(A, Wt, smA, smB, acc);

#pragma unroll
  for (int i = 0; i < 2; ++i)
#pragma unroll
    for (int j = 0; j < 2; ++j) {
      const int n = nbase + j * 16 + l16;
      const float bv = bias[n];
#pragma unroll
      for (int r = 0; r < 8; ++r) {
        const int m = mbase + i * 16 + hf * 8 + r;
        float vv = acc[i][j][r] + bv;
        vv = vv > 0.f ? vv : 0.f;
        Out[(size_t)m * Hc + n] = (f16)vv;
      }
    }
}

// ---------------------------------------------------------------------------
// GEMM 2: x += A @ Wt^T + bias (residual, f32 in-place at row stride 2H),
// scatter-store xv into all_layers[l, s, b, side*H + n].
// ---------------------------------------------------------------------------
__global__ __launch_bounds__(256)
void gemm_res_kernel(const f16* __restrict__ A, const f16* __restrict__ Wt,
                     const float* __restrict__ bias,
                     float* __restrict__ xbase,    // + side*H, row stride 2H
                     float* __restrict__ o0base)   // + l*S*B*2H + side*H
{
  __shared__ __align__(16) f16 smA[2 * ASZ];
  __shared__ __align__(16) f16 smB[2 * BSZ];

  const int lane  = threadIdx.x & 31;
  const int wave  = threadIdx.x >> 5;
  const int mbase = blockIdx.y * 128 + (wave & 3) * 32;
  const int nbase = blockIdx.x * 64  + (wave >> 2) * 32;
  const int hf  = lane >> 4;
  const int l16 = lane & 15;

  v8f_t acc[2][2] = {};
  gemm_mainloop(A, Wt, smA, smB, acc);

#pragma unroll
  for (int i = 0; i < 2; ++i)
#pragma unroll
    for (int j = 0; j < 2; ++j) {
      const int n = nbase + j * 16 + l16;
      const float bv = bias[n];
#pragma unroll
      for (int r = 0; r < 8; ++r) {
        const int m  = mbase + i * 16 + hf * 8 + r;   // m = b*S + s
        const int bb = m >> 10;
        const int ss = m & (Sc - 1);
        const size_t xoff = (size_t)m * (2 * Hc) + n;
        const float xv = xbase[xoff] + acc[i][j][r] + bv;
        xbase[xoff] = xv;
        o0base[((size_t)ss * Bc + bb) * (2 * Hc) + n] = xv;
      }
    }
}

// ---------------------------------------------------------------------------
// Host-side orchestration
// ---------------------------------------------------------------------------
extern "C" void kernel_launch(void* const* d_in, const int* in_sizes, int n_in,
                              void* d_out, int out_size, void* d_ws, size_t ws_size,
                              hipStream_t stream)
{
  (void)in_sizes; (void)n_in; (void)out_size; (void)ws_size;

  const float* inputs = (const float*)d_in[0];
  const float* lp  = (const float*)d_in[1];
  const float* rp  = (const float*)d_in[2];
  const float* lwt = (const float*)d_in[3];
  const float* rwt = (const float*)d_in[4];
  const float* lw1 = (const float*)d_in[5];
  const float* lb1 = (const float*)d_in[6];
  const float* lw2 = (const float*)d_in[7];
  const float* lb2 = (const float*)d_in[8];
  const float* lg  = (const float*)d_in[9];
  const float* lbt = (const float*)d_in[10];
  const float* rw1 = (const float*)d_in[11];
  const float* rb1 = (const float*)d_in[12];
  const float* rw2 = (const float*)d_in[13];
  const float* rb2 = (const float*)d_in[14];
  const float* rg  = (const float*)d_in[15];
  const float* rbt = (const float*)d_in[16];

  float* out0 = (float*)d_out;                              // all_layers [L,S,B,2H]
  float* out1 = out0 + (size_t)Lc * Sc * Bc * 2 * Hc;       // final x   [B,S,2H]

  // Workspace: [0,4MiB) f16 weights, [4,36MiB) LN out, [36,68MiB) relu out
  f16* w16 = (f16*)d_ws;
  f16* yh  = (f16*)((char*)d_ws + ((size_t)4  << 20));
  f16* h1h = (f16*)((char*)d_ws + ((size_t)36 << 20));

  wconv_kernel<<<2048, 256, 0, stream>>>(lw1, lw2, rw1, rw2, w16);
  window_kernel<<<16384, 256, 0, stream>>>(inputs, lp, rp, lwt, rwt, out1);

  const dim3 ggrid(Hc / 64, Mtot / 128);  // (8, 256)
  for (int l = 0; l < Lc; ++l) {
    for (int side = 0; side < 2; ++side) {
      const float* g  = (side ? rg  : lg)  + (size_t)l * Hc;
      const float* bt = (side ? rbt : lbt) + (size_t)l * Hc;
      const float* b1 = (side ? rb1 : lb1) + (size_t)l * Hc;
      const float* b2 = (side ? rb2 : lb2) + (size_t)l * Hc;
      const f16* w1 = w16 + (size_t)(side * 4 + l)     * HH;
      const f16* w2 = w16 + (size_t)(side * 4 + 2 + l) * HH;
      float* xb = out1 + (size_t)side * Hc;
      float* o0 = out0 + (size_t)l * Sc * Bc * 2 * Hc + (size_t)side * Hc;

      ln_kernel<<<Mtot / 8, 256, 0, stream>>>(xb, g, bt, yh);
      gemm_relu_kernel<<<ggrid, 256, 0, stream>>>(yh, w1, b1, h1h);
      gemm_res_kernel<<<ggrid, 256, 0, stream>>>(h1h, w2, b2, xb, o0);
    }
  }
}
